// BiCrossAttention_18820546691260
// MI455X (gfx1250) — matrix operand order repaired and checked by reading, not compiled, not run on hardware
//
#include <hip/hip_runtime.h>
#include <hip/hip_bf16.h>

// ---------------------------------------------------------------------------
// BiCrossAttention for MI455X (gfx1250): bf16 WMMA (16x16x32, f32 accum)
// + async global->LDS staging (ASYNCcnt path) for shared tiles
// ---------------------------------------------------------------------------
typedef __bf16 bf16;
typedef __attribute__((ext_vector_type(8)))  __bf16 v8bf;
typedef __attribute__((ext_vector_type(16))) __bf16 v16bf;
typedef __attribute__((ext_vector_type(8)))  float  v8f;

#define DIMD   512
#define BATCH  128
#define TE_    600
#define TF_    120
#define MAXPADK 608   // ceil(600/32)*32

static __device__ __forceinline__ v16bf cat8(v8bf lo, v8bf hi) {
  return __builtin_shufflevector(lo, hi, 0,1,2,3,4,5,6,7,8,9,10,11,12,13,14,15);
}
static __device__ __forceinline__ int imin(int a, int b){ return a < b ? a : b; }

// ---------------------------------------------------------------------------
// Async global -> LDS 16-byte copy (CDNA5 GLOBAL_LOAD_ASYNC_TO_LDS_B128).
// Inline asm: bypasses the builtin's HIP address-space Sema restrictions.
// vdst = LDS byte address (low 32 bits of the generic LDS pointer),
// vaddr = 64-bit global address, saddr = off.
// ---------------------------------------------------------------------------
static __device__ __forceinline__ void async_copy16(const bf16* g, bf16* l) {
  unsigned lds_off = (unsigned)(unsigned long long)l;
  asm volatile("global_load_async_to_lds_b128 %0, %1, off"
               :: "v"(lds_off), "v"(g) : "memory");
}

static __device__ __forceinline__ void wait_async0() {
#if __has_builtin(__builtin_amdgcn_s_wait_asynccnt)
  __builtin_amdgcn_s_wait_asynccnt(0);
#else
  asm volatile("s_wait_asynccnt 0x0" ::: "memory");
#endif
}

// A-fragment (16x32 bf16, row-major source, per-lane row pointer `pa`):
//   lanes 0-15: rows 0-15, elems 0-7 -> K=0..7,   elems 8-15 -> K=16..23
//   lanes 16-31: same rows,  elems 0-7 -> K=8..15, elems 8-15 -> K=24..31
static __device__ __forceinline__ v16bf load_a_frag(const bf16* pa, int h) {
  v8bf lo = *(const v8bf*)(pa + h * 8);
  v8bf hi = *(const v8bf*)(pa + 16 + h * 8);
  return cat8(lo, hi);
}
// B-fragment (32x16 bf16), column-contiguous source (`pb` points at K=0 of
// this lane's column): elems 0-7 -> K = h*8 + i, elems 8-15 -> K = 16 + h*8 + (i-8)
static __device__ __forceinline__ v16bf load_b_frag(const bf16* pb, int h) {
  v8bf lo = *(const v8bf*)(pb + h * 8);
  v8bf hi = *(const v8bf*)(pb + 16 + h * 8);
  return cat8(lo, hi);
}

// ---------------------------------------------------------------------------
// Elementwise f32 -> bf16
// ---------------------------------------------------------------------------
__global__ void cvt_bf16_kernel(const float* __restrict__ in,
                                bf16* __restrict__ out, size_t n) {
  size_t i = (size_t)blockIdx.x * blockDim.x + threadIdx.x;
  if (i < n) out[i] = (bf16)in[i];
}

// W[k][n] (512x512 f32) -> WT[n][k] bf16 (transposed for contiguous B frags)
__global__ void cvt_wt_kernel(const float* __restrict__ W,
                              bf16* __restrict__ WT) {
  int i = blockIdx.x * blockDim.x + threadIdx.x;   // i = n*512 + k
  int n = i >> 9, k = i & 511;
  WT[i] = (bf16)W[k * DIMD + n];
}

// ---------------------------------------------------------------------------
// Projection GEMM: out[M x 512] = X[M x 512] @ W + bias, bf16 in/out, f32 acc
// block = 128 threads (4 waves); block tile 64x64; wave tile 16x64.
// Weight tile (shared by all 4 waves) is double-buffered in LDS via
// async global->LDS loads; B fragments come from LDS (ds_load_b128).
// transpose_out: store per-batch transposed [D][T] (for V matrices).
// ---------------------------------------------------------------------------
__global__ void proj_gemm_kernel(const bf16* __restrict__ X,
                                 const bf16* __restrict__ WT,
                                 const float* __restrict__ bias,
                                 bf16* __restrict__ out,
                                 int M, int T, int transpose_out) {
  __shared__ __align__(16) bf16 Bs[2][64 * 32];   // [buf][n_local*32 + k]

  const int tid  = threadIdx.x;
  const int lane = tid & 31;
  const int wave = tid >> 5;
  const int m    = lane & 15;
  const int h    = lane >> 4;

  const int r0 = blockIdx.x * 64 + wave * 16;  // row base of this wave
  const int c0 = blockIdx.y * 64;              // col base of this block

  v8f acc[4];
  #pragma unroll
  for (int nt = 0; nt < 4; ++nt) acc[nt] = (v8f){0,0,0,0,0,0,0,0};

  const bf16* paRow = X + (size_t)(r0 + m) * DIMD;

  // Stage a 64(cols) x 32(k) weight tile into LDS buffer `buf`.
  // 256 b128 chunks, 2 per thread.
  auto stage = [&](int buf, int k0) {
    #pragma unroll
    for (int c = tid; c < 256; c += 128) {
      const int nl = c >> 2;            // local col 0..63
      const int k8 = (c & 3) << 3;      // k sub-offset 0,8,16,24
      async_copy16(WT + (size_t)(c0 + nl) * DIMD + k0 + k8,
                   &Bs[buf][nl * 32 + k8]);
    }
  };

  stage(0, 0);
  wait_async0();
  __syncthreads();

  for (int k0 = 0; k0 < DIMD; k0 += 32) {
    const int cur = (k0 >> 5) & 1;
    if (k0 + 32 < DIMD) stage(cur ^ 1, k0 + 32);

    __builtin_prefetch((const void*)(paRow + k0 + 64), 0, 1);
    v16bf a = load_a_frag(paRow + k0, h);
    #pragma unroll
    for (int nt = 0; nt < 4; ++nt) {
      const bf16* pb = &Bs[cur][((lane & 15) + nt * 16) * 32];
      v16bf b = load_b_frag(pb, h);
      acc[nt] = __builtin_amdgcn_wmma_f32_16x16x32_bf16(
          false, a, false, b, (short)0, acc[nt], false, false);
    }

    if (k0 + 32 < DIMD) {
      wait_async0();
      __syncthreads();
    }
  }

  #pragma unroll
  for (int nt = 0; nt < 4; ++nt) {
    const int col = c0 + nt * 16 + (lane & 15);
    const float bv = bias[col];
    #pragma unroll
    for (int rr = 0; rr < 8; ++rr) {
      const int row = r0 + rr + 8 * h;
      const float v = acc[nt][rr] + bv;
      if (transpose_out) {
        const int bidx = row / T;
        const int t    = row - bidx * T;
        out[((size_t)bidx * DIMD + col) * T + t] = (bf16)v;
      } else {
        out[(size_t)row * DIMD + col] = (bf16)v;
      }
    }
  }
}

// ---------------------------------------------------------------------------
// Fused masked cross-attention, one (batch, 16-query tile) per block.
// Q: [B][Tq][512] bf16 row-major; K: [B][Tk][512] bf16 row-major;
// VT: [B][512][Tk] bf16 (per-batch transposed); O: [B][Tq][512] f32.
// Q tile (shared by all waves) staged into LDS via async loads.
// dir==0 (e2f): allowed = key j in [q/20+20, q/20+80]
// dir==1 (f2e): allowed = q in [key/20+20, key/20+80]
// ---------------------------------------------------------------------------
__global__ void cross_attn_kernel(const bf16* __restrict__ Q,
                                  const bf16* __restrict__ K,
                                  const bf16* __restrict__ VT,
                                  float* __restrict__ O,
                                  int Tq, int Tk, int dir) {
  __shared__ __align__(16) bf16  Qs[16 * DIMD];     // 16 KB
  __shared__ __align__(16) float Sm[16 * MAXPADK];  // 38 KB
  __shared__ __align__(16) bf16  Pm[16 * MAXPADK];  // 19 KB
  __shared__ float rowsum[16];

  const int tid  = threadIdx.x;
  const int lane = tid & 31;
  const int wave = tid >> 5;
  const int m    = lane & 15;
  const int h    = lane >> 4;
  const int b    = blockIdx.y;
  const int qt   = blockIdx.x;

  const int PADK   = ((Tk + 31) / 32) * 32;
  const int ktiles = PADK / 16;
  const float scale = 0.044194173824159216f;   // 512^-0.5

  const bf16* Qb  = Q  + (size_t)b * Tq * DIMD;
  const bf16* Kb  = K  + (size_t)b * Tk * DIMD;
  const bf16* VTb = VT + (size_t)b * DIMD * Tk;

  // ---- Stage Q tile (16 rows x 512) into LDS: 1024 b128 chunks ----
  for (int c = tid; c < 1024; c += 128) {
    const int r   = c >> 6;              // row 0..15
    const int c16 = (c & 63) << 3;       // col 0..504 step 8
    const int qr  = imin(qt * 16 + r, Tq - 1);
    async_copy16(Qb + (size_t)qr * DIMD + c16, &Qs[r * DIMD + c16]);
  }
  wait_async0();
  __syncthreads();

  // ---- Phase 1: S = scale * Q Kt  (waves stride over key tiles) ----
  const bf16* paRow = &Qs[m * DIMD];

  for (int kt = wave; kt < ktiles; kt += 4) {
    const int key = imin(kt * 16 + (lane & 15), Tk - 1);
    const bf16* pbCol = Kb + (size_t)key * DIMD;
    v8f acc = (v8f){0,0,0,0,0,0,0,0};
    for (int dk = 0; dk < DIMD; dk += 32) {
      v16bf a = load_a_frag(paRow + dk, h);
      v16bf bfr = load_b_frag(pbCol + dk, h);
      acc = __builtin_amdgcn_wmma_f32_16x16x32_bf16(
          false, a, false, bfr, (short)0, acc, false, false);
    }
    #pragma unroll
    for (int rr = 0; rr < 8; ++rr) {
      Sm[(rr + 8 * h) * PADK + kt * 16 + (lane & 15)] = acc[rr] * scale;
    }
  }
  __syncthreads();

  // ---- Phase 2: masked softmax per row (reference semantics: -1e9) ----
  if (tid < 16) {
    const int row = tid;
    const int q   = qt * 16 + row;
    float mx = -1e30f;
    for (int j = 0; j < Tk; ++j) {
      bool ok;
      if (dir == 0) { int j0 = q / 20; ok = (j >= j0 + 20) && (j <= j0 + 80); }
      else          { int j0 = j / 20; ok = (q >= j0 + 20) && (q <= j0 + 80); }
      float x = ok ? Sm[row * PADK + j] : -1e9f;
      mx = fmaxf(mx, x);
    }
    float sum = 0.0f;
    for (int j = 0; j < PADK; ++j) {
      float p = 0.0f;
      if (j < Tk) {
        bool ok;
        if (dir == 0) { int j0 = q / 20; ok = (j >= j0 + 20) && (j <= j0 + 80); }
        else          { int j0 = j / 20; ok = (q >= j0 + 20) && (q <= j0 + 80); }
        float x = ok ? Sm[row * PADK + j] : -1e9f;
        p = __expf(x - mx);
        sum += p;
      }
      Pm[row * PADK + j] = (bf16)p;
    }
    rowsum[row] = (sum > 0.0f) ? (1.0f / sum) : 0.0f;
  }
  __syncthreads();

  // ---- Phase 3: O = (P V) * (1/rowsum); wave w owns d-range [w*128, +128) ----
  v8f o[8];
  #pragma unroll
  for (int nt = 0; nt < 8; ++nt) o[nt] = (v8f){0,0,0,0,0,0,0,0};

  const bf16* pRow = &Pm[m * PADK];
  for (int ks = 0; ks < PADK; ks += 32) {
    v16bf a = load_a_frag(pRow + ks, h);
    #pragma unroll
    for (int nt = 0; nt < 8; ++nt) {
      const int d = wave * 128 + nt * 16 + (lane & 15);
      const bf16* pv = VTb + (size_t)d * Tk + ks;   // contiguous over kseq
      v16bf bfr = load_b_frag(pv, h);
      o[nt] = __builtin_amdgcn_wmma_f32_16x16x32_bf16(
          false, a, false, bfr, (short)0, o[nt], false, false);
    }
  }

  #pragma unroll
  for (int nt = 0; nt < 8; ++nt) {
    #pragma unroll
    for (int rr = 0; rr < 8; ++rr) {
      const int rowm = rr + 8 * h;
      const int q    = qt * 16 + rowm;
      if (q < Tq) {
        const int col = wave * 128 + nt * 16 + (lane & 15);
        O[((size_t)b * Tq + q) * DIMD + col] = o[nt][rr] * rowsum[rowm];
      }
    }
  }
}

// ---------------------------------------------------------------------------
// Host-side orchestration
// ---------------------------------------------------------------------------
extern "C" void kernel_launch(void* const* d_in, const int* in_sizes, int n_in,
                              void* d_out, int out_size, void* d_ws, size_t ws_size,
                              hipStream_t stream) {
  (void)in_sizes; (void)n_in; (void)out_size; (void)ws_size;

  const float* eeg   = (const float*)d_in[0];
  const float* fnirs = (const float*)d_in[1];
  const float* W[6]  = {(const float*)d_in[2],  (const float*)d_in[4],
                        (const float*)d_in[6],  (const float*)d_in[8],
                        (const float*)d_in[10], (const float*)d_in[12]};
  const float* bias[6] = {(const float*)d_in[3],  (const float*)d_in[5],
                          (const float*)d_in[7],  (const float*)d_in[9],
                          (const float*)d_in[11], (const float*)d_in[13]};
  // order: 0=Wqe 1=Wke 2=Wve 3=Wqf 4=Wkf 5=Wvf

  const size_t eegN  = (size_t)BATCH * TE_ * DIMD;   // 39,321,600
  const size_t fnirN = (size_t)BATCH * TF_ * DIMD;   //  7,864,320
  const size_t wN    = (size_t)DIMD * DIMD;

  char* base = (char*)d_ws;
  size_t off = 0;
  auto alloc_bf = [&](size_t elems) -> bf16* {
    bf16* p = (bf16*)(base + off);
    off += ((elems * sizeof(bf16) + 255) & ~(size_t)255) + 256;  // slack+align
    return p;
  };
  bf16* eeg_bf  = alloc_bf(eegN);
  bf16* fnir_bf = alloc_bf(fnirN);
  bf16* WT[6];
  for (int i = 0; i < 6; ++i) WT[i] = alloc_bf(wN);
  bf16* Qe  = alloc_bf(eegN);
  bf16* Ke  = alloc_bf(eegN);
  bf16* VeT = alloc_bf(eegN);
  bf16* Qf  = alloc_bf(fnirN);
  bf16* Kf  = alloc_bf(fnirN);
  bf16* VfT = alloc_bf(fnirN);

  float* aligned_eeg   = (float*)d_out;                               // [B,Tf,D]
  float* aligned_fnirs = (float*)d_out + (size_t)BATCH * TF_ * DIMD;  // [B,Te,D]

  // 1) conversions
  cvt_bf16_kernel<<<(int)((eegN  + 255) / 256), 256, 0, stream>>>(eeg,   eeg_bf,  eegN);
  cvt_bf16_kernel<<<(int)((fnirN + 255) / 256), 256, 0, stream>>>(fnirs, fnir_bf, fnirN);
  for (int i = 0; i < 6; ++i)
    cvt_wt_kernel<<<(int)(wN / 256), 256, 0, stream>>>(W[i], WT[i]);

  // 2) projections (M is a multiple of 64 in both cases)
  const int Me = BATCH * TE_, Mf = BATCH * TF_;
  dim3 blk(128);
  dim3 gE(Me / 64, DIMD / 64), gF(Mf / 64, DIMD / 64);
  proj_gemm_kernel<<<gE, blk, 0, stream>>>(eeg_bf,  WT[0], bias[0], Qe,  Me, TE_, 0);
  proj_gemm_kernel<<<gE, blk, 0, stream>>>(eeg_bf,  WT[1], bias[1], Ke,  Me, TE_, 0);
  proj_gemm_kernel<<<gE, blk, 0, stream>>>(eeg_bf,  WT[2], bias[2], VeT, Me, TE_, 1);
  proj_gemm_kernel<<<gF, blk, 0, stream>>>(fnir_bf, WT[3], bias[3], Qf,  Mf, TF_, 0);
  proj_gemm_kernel<<<gF, blk, 0, stream>>>(fnir_bf, WT[4], bias[4], Kf,  Mf, TF_, 0);
  proj_gemm_kernel<<<gF, blk, 0, stream>>>(fnir_bf, WT[5], bias[5], VfT, Mf, TF_, 1);

  // 3) attention
  // aligned_fnirs = attend(Qe, Kf, Vf, mask_e2f): Tq=600, Tk=120, dir=0
  dim3 gA((TE_ + 15) / 16, BATCH);
  cross_attn_kernel<<<gA, blk, 0, stream>>>(Qe, Kf, VfT, aligned_fnirs, TE_, TF_, 0);
  // aligned_eeg = attend(Qf, Ke, Ve, mask_f2e): Tq=120, Tk=600, dir=1
  dim3 gB((TF_ + 15) / 16, BATCH);
  cross_attn_kernel<<<gB, blk, 0, stream>>>(Qf, Ke, VeT, aligned_eeg, TF_, TE_, 1);
}
